// SignAdaptorV2_53017076302453
// MI455X (gfx1250) — matrix operand order repaired and checked by reading, not compiled, not run on hardware
//
#include <hip/hip_runtime.h>
#include <hip/hip_bf16.h>

// ---------------------------------------------------------------------------
// SignAdaptor for MI455X (gfx1250, wave32, WMMA + TDM).
//   sp = image @ W_sp.T + b_sp ; mp = clip @ W_mo.T + b_mo ; ragged scatter.
// bf16 WMMA (f32 accumulate): 63 GFLOP vs ~227 MB moved needs the 16-bit
// matrix pipe to reach the 23.3 TB/s roofline. Weights staged per workgroup
// into LDS by the Tensor Data Mover (double-buffered, TENSORcnt), shared by
// 8 waves; each wave owns 2 M-tiles so every B fetch feeds 2 WMMAs.
// ---------------------------------------------------------------------------

typedef __attribute__((ext_vector_type(16))) __bf16 v16bf;
typedef __attribute__((ext_vector_type(8)))  __bf16 v8bf;
typedef __attribute__((ext_vector_type(8)))  float  v8f;
typedef __attribute__((ext_vector_type(4)))  unsigned int u32x4;
typedef __attribute__((ext_vector_type(8)))  int i32x8;
typedef __attribute__((ext_vector_type(4)))  int i32x4;

// ---------------- fp32 -> bf16 (round-to-nearest-even) ---------------------
__global__ void cvt_f32_bf16_kernel(const float* __restrict__ in,
                                    unsigned short* __restrict__ out,
                                    long long n) {
    long long i = (long long)blockIdx.x * blockDim.x + threadIdx.x;
    if (i < n) {
        unsigned int u = __float_as_uint(in[i]);
        u += 0x7fffu + ((u >> 16) & 1u);     // RNE
        out[i] = (unsigned short)(u >> 16);
    }
}

// ---------------- pad-fill of x + src_length tail --------------------------
__global__ void fill_pad_kernel(float* __restrict__ x, long long n4,
                                const int* __restrict__ pad,
                                int* __restrict__ slen,
                                const int* __restrict__ nf,
                                const int* __restrict__ nc, int Bn) {
    long long i = (long long)blockIdx.x * blockDim.x + threadIdx.x;
    float pv = (float)pad[0];
    if (i < n4) {
        float4 v = make_float4(pv, pv, pv, pv);
        ((float4*)x)[i] = v;
    }
    if (i < Bn) slen[i] = nf[i] + nc[i];
}

// ---------------- per-row scatter destination (element offset) -------------
__global__ void rowdest_kernel(const int* __restrict__ nf,
                               const int* __restrict__ nc,
                               int max_len, int F_tot, int C_tot, int Bn,
                               int* __restrict__ rd_f, int* __restrict__ rd_c) {
    const int LLM = 4096;
    int g = blockIdx.x * blockDim.x + threadIdx.x;
    if (g < F_tot) {
        int s = -1, start = 0, cum = 0;
        for (int i = 0; i < Bn; ++i) {
            int e = cum + nf[i];
            if (s < 0 && g < e) { s = i; start = cum; }
            cum = e;
        }
        rd_f[g] = (s * max_len + (g - start)) * LLM;
    } else if (g < F_tot + C_tot) {
        int gc = g - F_tot;
        int s = -1, start = 0, cum = 0;
        for (int i = 0; i < Bn; ++i) {
            int e = cum + nc[i];
            if (s < 0 && gc < e) { s = i; start = cum; }
            cum = e;
        }
        rd_c[gc] = (s * max_len + nf[s] + (gc - start)) * LLM;
    }
}

// ---------------- TDM helper: B tile (64 W-rows x KC K-elems) -> LDS -------
// 2D descriptor: dim0 = K (contiguous, stride 1024 elems between W rows),
// tile = KC x 64, data_size = 2B. LDS pad: +4 dwords every 64 dwords
// (pad_interval code 5, pad_amount code 3) -> column stride 272B,
// bank stride 68 % 64 = 4 -> conflict-free ds_load_b128, 16B aligned.
#define KC         128
#define COLSTRIDE  272            // (KC*2 bytes) + 16B pad
#define BUFBYTES   (64 * COLSTRIDE)

__device__ __forceinline__ void tdm_load_b_tile(const unsigned short* wsrc,
                                                unsigned lds_byte_off) {
    unsigned long long ga = (unsigned long long)(const void*)wsrc;
    u32x4 g0;
    g0[0] = 1u;                                            // count=1, user mode
    g0[1] = (unsigned)__builtin_amdgcn_readfirstlane((int)lds_byte_off);
    g0[2] = (unsigned)__builtin_amdgcn_readfirstlane((int)(unsigned)ga);
    g0[3] = ((unsigned)__builtin_amdgcn_readfirstlane((int)(unsigned)(ga >> 32))
             & 0x01FFFFFFu) | 0x80000000u;                 // addr[56:32] | type=2
    i32x8 g1;
    g1[0] = 0x07510000;  // data_size=2B, pad_enable, interval=64dw, amount=4dw
    g1[1] = 0;                           // tensor_dim0 = 1<<20 (lo16 = 0)
    g1[2] = 0x10 | (0 << 16);            // tensor_dim0 hi16, tensor_dim1 lo16
    g1[3] = 0x10 | (KC << 16);           // tensor_dim1 hi16, tile_dim0 = KC
    g1[4] = 64;                          // tile_dim1 = 64, tile_dim2 = 0
    g1[5] = 1024;                        // tensor_dim0_stride = 1024 elems
    g1[6] = 0;
    g1[7] = 0;
    i32x4 gz = {0, 0, 0, 0};
#if __clang_major__ >= 23
    i32x8 gz8 = {0, 0, 0, 0, 0, 0, 0, 0};
    __builtin_amdgcn_tensor_load_to_lds(g0, g1, gz, gz, gz8, 0);
#else
    __builtin_amdgcn_tensor_load_to_lds(g0, g1, gz, gz, 0);
#endif
}

// ---------------- WMMA GEMM + scatter --------------------------------------
// Block = 8 waves = 256 M rows x 64 N cols. Wave w owns row tiles
// [base+w*16) and [base+128+w*16). B tile staged in LDS by TDM (wave 0),
// double-buffered over K chunks of KC. Per k-step: 4 global b128 (A),
// 8 ds b128 (B, 4 distinct register sets -> incremental DScnt waits),
// 8 WMMAs. 256 WMMAs per wave.
__global__ void __launch_bounds__(256)
wmma_gemm_scatter_kernel(const unsigned short* __restrict__ Abf,
                         const unsigned short* __restrict__ Wbf,
                         const float* __restrict__ bias,
                         const int* __restrict__ rowdest,
                         float* __restrict__ xout,
                         int R) {
    constexpr int K = 1024;
    constexpr int LLM = 4096;
    constexpr int NT = 4;                 // 4 N-subtiles of 16 per wave
    constexpr int MT = 2;                 // 2 M-tiles of 16 per wave
    constexpr int NCHUNK = K / KC;        // 8

    __shared__ unsigned char ldsb[2 * BUFBYTES];
    const unsigned lds_base =
        (unsigned)(unsigned long long)(const void*)&ldsb[0];

    int tid  = threadIdx.x;
    int wave = tid >> 5;
    int lane = tid & 31;
    int half = lane >> 4;
    int mrow = lane & 15;

    int blk_m  = (blockIdx.x / (LLM / 64)) * 256;
    int n_base = (blockIdx.x % (LLM / 64)) * 64;
    int m0 = blk_m + wave * 16;           // M-tile 0
    int m1 = m0 + 128;                    // M-tile 1

    int ar0 = m0 + mrow; if (ar0 >= R) ar0 = R - 1;   // clamp; stores guarded
    int ar1 = m1 + mrow; if (ar1 >= R) ar1 = R - 1;
    const __bf16* aptr0 = (const __bf16*)Abf + (size_t)ar0 * K;
    const __bf16* aptr1 = (const __bf16*)Abf + (size_t)ar1 * K;

    v8f acc[MT][NT];
#pragma unroll
    for (int nt = 0; nt < NT; ++nt) {
        float bv = bias[n_base + nt * 16 + mrow];
#pragma unroll
        for (int mt = 0; mt < MT; ++mt)
#pragma unroll
            for (int i = 0; i < 8; ++i) acc[mt][nt][i] = bv;
    }

    union BU { v16bf v; v8bf h[2]; };

    const unsigned short* wblock = Wbf + (size_t)n_base * K;

    // prime chunk 0 into buffer 0
    if (tid < 32) {
        tdm_load_b_tile(wblock, lds_base);
        __builtin_amdgcn_s_wait_tensorcnt(0);
    }
    __syncthreads();

    for (int chunk = 0; chunk < NCHUNK; ++chunk) {
        // prefetch next K chunk into the other buffer (TDM, async)
        if (tid < 32 && chunk + 1 < NCHUNK)
            tdm_load_b_tile(wblock + (chunk + 1) * KC,
                            lds_base + ((chunk + 1) & 1) * BUFBYTES);

        const unsigned char* lbuf = &ldsb[(chunk & 1) * BUFBYTES];
        const int kglob = chunk * KC;

#pragma unroll
        for (int kk = 0; kk < KC; kk += 32) {
            // A operands: two M-tiles, per lane two 16B K-runs (8*half, 16+8*half) each
            BU a0, a1;
            a0.h[0] = *(const v8bf*)(aptr0 + kglob + kk + half * 8);
            a0.h[1] = *(const v8bf*)(aptr0 + kglob + kk + 16 + half * 8);
            a1.h[0] = *(const v8bf*)(aptr1 + kglob + kk + half * 8);
            a1.h[1] = *(const v8bf*)(aptr1 + kglob + kk + 16 + half * 8);
            __builtin_prefetch(aptr0 + kglob + kk + 256, 0, 1);  // stream A
            __builtin_prefetch(aptr1 + kglob + kk + 256, 0, 1);

            // B operands: all 4 subtiles in distinct registers first
            BU b[NT];
#pragma unroll
            for (int nt = 0; nt < NT; ++nt) {
                const unsigned char* colp =
                    lbuf + (nt * 16 + mrow) * COLSTRIDE + (kk + half * 8) * 2;
                b[nt].h[0] = *(const v8bf*)(colp);
                b[nt].h[1] = *(const v8bf*)(colp + 32);
            }

            // 8 WMMAs: each B reused by both M-tiles
#pragma unroll
            for (int nt = 0; nt < NT; ++nt) {
                acc[0][nt] = __builtin_amdgcn_wmma_f32_16x16x32_bf16(
                    false, a0.v, false, b[nt].v, (short)0, acc[0][nt],
                    false, false);
                acc[1][nt] = __builtin_amdgcn_wmma_f32_16x16x32_bf16(
                    false, a1.v, false, b[nt].v, (short)0, acc[1][nt],
                    false, false);
            }
        }

        // publish the prefetched buffer: TDM done (wave 0) -> barrier (all)
        if (tid < 32) __builtin_amdgcn_s_wait_tensorcnt(0);
        __syncthreads();
    }

    // C/D layout: VGPR r -> M = r + 8*half, N = mrow (+ n-subtile offset)
#pragma unroll
    for (int mt = 0; mt < MT; ++mt) {
        int mb = (mt == 0) ? m0 : m1;
#pragma unroll
        for (int r = 0; r < 8; ++r) {
            int m = mb + r + half * 8;
            if (m < R) {
                float* dst = xout + rowdest[m] + n_base + mrow;
#pragma unroll
                for (int nt = 0; nt < NT; ++nt) dst[nt * 16] = acc[mt][nt][r];
            }
        }
    }
}

// ---------------------------------------------------------------------------
extern "C" void kernel_launch(void* const* d_in, const int* in_sizes, int n_in,
                              void* d_out, int out_size, void* d_ws, size_t ws_size,
                              hipStream_t stream) {
    const float* img  = (const float*)d_in[0];
    const float* clip = (const float*)d_in[1];
    const int*   nf   = (const int*)d_in[2];
    const int*   nc   = (const int*)d_in[3];
    const float* W_sp = (const float*)d_in[4];
    const float* b_sp = (const float*)d_in[5];
    const float* W_mo = (const float*)d_in[6];
    const float* b_mo = (const float*)d_in[7];
    const int*   pad  = (const int*)d_in[8];
    // d_in[9] = max_len on device; host value recovered from out_size.

    const int SP = 1024, MO = 1024, LLM = 4096;
    const int Bn = in_sizes[2];                       // 16 samples
    const int F_tot = in_sizes[0] / SP;
    const int C_tot = in_sizes[1] / MO;
    const int max_len = (out_size - Bn) / (Bn * LLM);
    const long long n_x = (long long)Bn * max_len * LLM;

    // ---- workspace carve-up (bf16 copies + row-destination tables) --------
    size_t off = 0;
    auto carve = [&](size_t bytes) {
        off = (off + 255) & ~(size_t)255;
        void* p = (char*)d_ws + off;
        off += bytes;
        return p;
    };
    unsigned short* Af  = (unsigned short*)carve((size_t)F_tot * SP * 2);
    unsigned short* Ac  = (unsigned short*)carve((size_t)C_tot * MO * 2);
    unsigned short* Wsb = (unsigned short*)carve((size_t)LLM * SP * 2);
    unsigned short* Wmb = (unsigned short*)carve((size_t)LLM * MO * 2);
    int* rd_f = (int*)carve((size_t)F_tot * 4);
    int* rd_c = (int*)carve((size_t)C_tot * 4);
    (void)ws_size; (void)n_in;

    // ---- 1) fp32 -> bf16 conversions --------------------------------------
    {
        long long n;
        n = (long long)F_tot * SP;
        cvt_f32_bf16_kernel<<<(unsigned)((n + 255) / 256), 256, 0, stream>>>(img, Af, n);
        n = (long long)C_tot * MO;
        cvt_f32_bf16_kernel<<<(unsigned)((n + 255) / 256), 256, 0, stream>>>(clip, Ac, n);
        n = (long long)LLM * SP;
        cvt_f32_bf16_kernel<<<(unsigned)((n + 255) / 256), 256, 0, stream>>>(W_sp, Wsb, n);
        n = (long long)LLM * MO;
        cvt_f32_bf16_kernel<<<(unsigned)((n + 255) / 256), 256, 0, stream>>>(W_mo, Wmb, n);
    }

    // ---- 2) pad-fill x and write src_length tail --------------------------
    {
        long long n4 = n_x / 4;
        int* slen = (int*)d_out + n_x;
        fill_pad_kernel<<<(unsigned)((n4 + 255) / 256), 256, 0, stream>>>(
            (float*)d_out, n4, pad, slen, nf, nc, Bn);
    }

    // ---- 3) ragged scatter row destinations -------------------------------
    {
        int n = F_tot + C_tot;
        rowdest_kernel<<<(n + 255) / 256, 256, 0, stream>>>(
            nf, nc, max_len, F_tot, C_tot, Bn, rd_f, rd_c);
    }

    // ---- 4) WMMA GEMMs (TDM-staged weights) with fused scatter ------------
    {
        int blocks = ((F_tot + 255) / 256) * (LLM / 64);
        wmma_gemm_scatter_kernel<<<blocks, 256, 0, stream>>>(
            Af, Wsb, b_sp, rd_f, (float*)d_out, F_tot);

        blocks = ((C_tot + 255) / 256) * (LLM / 64);
        wmma_gemm_scatter_kernel<<<blocks, 256, 0, stream>>>(
            Ac, Wmb, b_mo, rd_c, (float*)d_out, C_tot);
    }
}
